// ODConv_19327352832738
// MI455X (gfx1250) — matrix operand order repaired and checked by reading, not compiled, not run on hardware
//
#include <hip/hip_runtime.h>
#include <math.h>

typedef __attribute__((ext_vector_type(16))) __bf16 v16bf;
typedef __attribute__((ext_vector_type(8)))  float  v8f;

struct __align__(16) Q4 { unsigned int x, y, z, w; };
struct FragBits { Q4 lo, hi; };

static __device__ __forceinline__ v16bf frag_cast(const FragBits& f) {
    return __builtin_bit_cast(v16bf, f);
}

static __device__ __forceinline__ unsigned short f2bf(float f) {
    unsigned int u = __float_as_uint(f);
    u = (u + 0x7FFFu + ((u >> 16) & 1u)) >> 16;
    return (unsigned short)u;
}

#define BATCH 32
#define CIN 256
#define COUT 256
#define HH 56
#define WW 56
#define HP 58          // padded
#define ATT 16
#define NEXP 4

// ---------------------------------------------------------------------------
// K1: global average pool. One block per (b, c).
// ---------------------------------------------------------------------------
__global__ void pool_kernel(const float* __restrict__ x, float* __restrict__ pooled) {
    int bc = blockIdx.x;                       // b*256 + c
    const float* p = x + (size_t)bc * (HH * WW);
    float s = 0.f;
    for (int i = threadIdx.x; i < HH * WW; i += 256) s += p[i];
    __shared__ float red[8];
    int lane = threadIdx.x & 31, wid = threadIdx.x >> 5;
    s += __shfl_down(s, 16, 32);
    s += __shfl_down(s,  8, 32);
    s += __shfl_down(s,  4, 32);
    s += __shfl_down(s,  2, 32);
    s += __shfl_down(s,  1, 32);
    if (lane == 0) red[wid] = s;
    __syncthreads();
    if (threadIdx.x == 0) {
        float t = 0.f;
        for (int i = 0; i < 8; ++i) t += red[i];
        pooled[bc] = t * (1.f / (HH * WW));
    }
}

// ---------------------------------------------------------------------------
// K2: attention MLPs. One block per sample.
// ---------------------------------------------------------------------------
__global__ void attn_kernel(const float* __restrict__ pooled,
                            const float* __restrict__ fc_w,
                            const float* __restrict__ ch_w, const float* __restrict__ ch_b,
                            const float* __restrict__ fl_w, const float* __restrict__ fl_b,
                            const float* __restrict__ sp_w, const float* __restrict__ sp_b,
                            const float* __restrict__ ex_w, const float* __restrict__ ex_b,
                            float* __restrict__ ch, float* __restrict__ fl,
                            float* __restrict__ sp, float* __restrict__ ex) {
    int b = blockIdx.x, tid = threadIdx.x;
    __shared__ float z[ATT];
    __shared__ float lg[NEXP];
    const float* pr = pooled + b * CIN;
    if (tid < ATT) {
        float a = 0.f;
        for (int c = 0; c < CIN; ++c) a += pr[c] * fc_w[tid * CIN + c];
        z[tid] = a > 0.f ? a : 0.f;
    }
    __syncthreads();
    {   // channel (Cin) and filter (Cout) attentions, one per thread
        float ac = ch_b[tid], af = fl_b[tid];
        #pragma unroll
        for (int a = 0; a < ATT; ++a) {
            float za = z[a];
            ac += za * ch_w[tid * ATT + a];
            af += za * fl_w[tid * ATT + a];
        }
        ch[b * CIN + tid] = 1.f / (1.f + expf(-ac));
        fl[b * COUT + tid] = 1.f / (1.f + expf(-af));
    }
    if (tid < 9) {
        float a = sp_b[tid];
        #pragma unroll
        for (int k = 0; k < ATT; ++k) a += z[k] * sp_w[tid * ATT + k];
        sp[b * 9 + tid] = 1.f / (1.f + expf(-a));
    }
    if (tid < NEXP) {
        float a = ex_b[tid];
        #pragma unroll
        for (int k = 0; k < ATT; ++k) a += z[k] * ex_w[tid * ATT + k];
        lg[tid] = a;
    }
    __syncthreads();
    if (tid == 0) {
        float mx = fmaxf(fmaxf(lg[0], lg[1]), fmaxf(lg[2], lg[3]));
        float e0 = expf(lg[0] - mx), e1 = expf(lg[1] - mx);
        float e2 = expf(lg[2] - mx), e3 = expf(lg[3] - mx);
        float inv = 1.f / (e0 + e1 + e2 + e3);
        ex[b * 4 + 0] = e0 * inv; ex[b * 4 + 1] = e1 * inv;
        ex[b * 4 + 2] = e2 * inv; ex[b * 4 + 3] = e3 * inv;
    }
}

// ---------------------------------------------------------------------------
// K3: effective weights, bf16, layout [b][pq][o][i] (i contiguous).
// grid = (256 /*o*/, 32 /*b*/), block 256 (thread = i).
// ---------------------------------------------------------------------------
__global__ void weff_kernel(const float* __restrict__ weight,
                            const float* __restrict__ ch, const float* __restrict__ fl,
                            const float* __restrict__ sp, const float* __restrict__ ex,
                            unsigned short* __restrict__ weff) {
    int o = blockIdx.x, b = blockIdx.y, i = threadIdx.x;
    const size_t ESTR = (size_t)COUT * CIN * 9;
    float s = fl[b * COUT + o] * ch[b * CIN + i];
    float e0 = ex[b * 4 + 0], e1 = ex[b * 4 + 1], e2 = ex[b * 4 + 2], e3 = ex[b * 4 + 3];
    const float* w0 = weight + ((size_t)o * CIN + i) * 9;
    float acc[9];
    #pragma unroll
    for (int k = 0; k < 9; ++k)
        acc[k] = e0 * w0[k] + e1 * w0[k + ESTR] + e2 * w0[k + 2 * ESTR] + e3 * w0[k + 3 * ESTR];
    #pragma unroll
    for (int k = 0; k < 9; ++k)
        weff[(((size_t)b * 9 + k) * COUT + o) * CIN + i] = f2bf(s * sp[b * 9 + k] * acc[k]);
}

// ---------------------------------------------------------------------------
// K4: x -> padded channels-last bf16 [b][58][58][256] via LDS transpose.
// grid = (56 /*h*/, 4 /*c chunk of 64*/, 32 /*b*/), block 256.
// (border already zeroed by memset)
// ---------------------------------------------------------------------------
__global__ void xpad_kernel(const float* __restrict__ x, unsigned short* __restrict__ xcl) {
    int h = blockIdx.x, c0 = blockIdx.y * 64, b = blockIdx.z, tid = threadIdx.x;
    __shared__ unsigned short lt[WW * 65];
    for (int idx = tid; idx < 64 * WW; idx += 256) {
        int cc = idx / WW, w = idx - cc * WW;
        float v = x[(((size_t)b * CIN + c0 + cc) * HH + h) * WW + w];
        lt[w * 65 + cc] = f2bf(v);
    }
    __syncthreads();
    unsigned int* outp = (unsigned int*)xcl;
    for (int idx = tid; idx < WW * 32; idx += 256) {
        int w = idx >> 5, cp = idx & 31;
        unsigned int v = (unsigned int)lt[w * 65 + 2 * cp] |
                         ((unsigned int)lt[w * 65 + 2 * cp + 1] << 16);
        outp[((((size_t)b * HP + h + 1) * HP + (w + 1)) * CIN + c0) / 2 + cp] = v;
    }
}

// ---------------------------------------------------------------------------
// K5: main implicit-GEMM conv with bf16 WMMA.
// Block: 128 out-channels x 64 pixels (8x8 spatial). 8 waves, wave = 32x32.
// All 9 taps' weight tiles staged in LDS per 32-channel chunk -> only two
// barriers per chunk, 36 unfenced WMMAs per chunk per wave.
// grid = (49 /*7x7 spatial tiles*/, 2 /*M tiles*/, 32 /*b*/), block 256.
// ---------------------------------------------------------------------------
__global__ __launch_bounds__(256) void conv_kernel(const unsigned short* __restrict__ weff,
                                                   const unsigned short* __restrict__ xcl,
                                                   float* __restrict__ out) {
    __shared__ __align__(16) unsigned short wsl[9 * 128 * 40]; // 9 taps x 128 o x 32 k (pad 40)
    __shared__ __align__(16) unsigned short xsl[100 * 40];     // 10x10 pixels x 32 k (pad 40)

    int tid = threadIdx.x, lane = tid & 31;
    int wid = tid >> 5;
    int wave_m = wid & 3, wave_n = wid >> 2;
    int b = blockIdx.z, o0 = blockIdx.y * 128;
    int tr = blockIdx.x / 7, tc = blockIdx.x % 7;
    int H0 = tr * 8, W0 = tc * 8;

    int m  = lane & 15;           // A row / D column
    int hi = lane >> 4;           // lane half
    int pr = (lane & 15) >> 3;    // pixel row within 2x8 n-tile
    int pc = lane & 7;            // pixel col

    v8f acc[2][2] = {};

    const size_t xbase = (((size_t)b * HP + H0) * HP + W0) * CIN;  // slab origin (padded)
    const size_t wbase = (size_t)b * 9 * (COUT * CIN) + (size_t)o0 * CIN;

    // per-thread slab-load coords (400 uint4 transfers, 2 per thread)
    for (int cin0 = 0; cin0 < CIN; cin0 += 32) {
        __syncthreads();  // previous chunk finished reading LDS
        // stage 10x10 input slab for this channel chunk (reused by all 9 taps)
        for (int u = tid; u < 400; u += 256) {
            int pix = u >> 2, q4 = u & 3;
            int sr = pix / 10, sc = pix - sr * 10;
            *(Q4*)&xsl[pix * 40 + q4 * 8] =
                *(const Q4*)&xcl[xbase + ((size_t)sr * HP + sc) * CIN + cin0 + q4 * 8];
        }
        // stage all 9 taps' 128x32 weight tiles (4608 uint4, 18 per thread)
        for (int u = tid; u < 4608; u += 256) {
            int q4  = u & 3;
            int o   = (u >> 2) & 127;
            int tap = u >> 9;
            *(Q4*)&wsl[(tap * 128 + o) * 40 + q4 * 8] =
                *(const Q4*)&weff[wbase + (size_t)tap * (COUT * CIN) +
                                  (size_t)o * CIN + cin0 + q4 * 8];
        }
        if (cin0 + 32 < CIN) {   // prefetch next chunk's streams into GL2
            __builtin_prefetch(&weff[wbase + (size_t)tid * 2 * CIN + cin0 + 32], 0, 0);
            __builtin_prefetch(&xcl[xbase + ((size_t)(tid >> 5) * HP + (tid & 31) % 10) * CIN +
                                    cin0 + 32], 0, 0);
        }
        __syncthreads();  // slab + weights visible

        #pragma unroll
        for (int p = 0; p < 3; ++p) {
            #pragma unroll
            for (int q = 0; q < 3; ++q) {
                const int tap = p * 3 + q;
                // A fragments (16x32 bf16): lane m holds K pairs; VGPR v<4 -> K=2v+8*hi,
                // v>=4 -> K=16+2(v-4)+8*hi  => elems [0..7]=K 8hi.., [8..15]=K 16+8hi..
                FragBits fa0, fa1, fb0, fb1;
                const unsigned short* ab = &wsl[(tap * 128 + wave_m * 32 + m) * 40 + hi * 8];
                fa0.lo = *(const Q4*)ab;
                fa0.hi = *(const Q4*)(ab + 16);
                fa1.lo = *(const Q4*)(ab + 640);        // +16 rows
                fa1.hi = *(const Q4*)(ab + 656);
                // B fragments (32x16 bf16): lane n holds col, elems [0..7]=K 8hi.., [8..15]=K 16+8hi..
                const unsigned short* bb =
                    &xsl[((wave_n * 4 + pr + p) * 10 + (pc + q)) * 40 + hi * 8];
                fb0.lo = *(const Q4*)bb;
                fb0.hi = *(const Q4*)(bb + 16);
                fb1.lo = *(const Q4*)(bb + 800);        // +2 pixel rows
                fb1.hi = *(const Q4*)(bb + 816);

                v16bf a0 = frag_cast(fa0), a1 = frag_cast(fa1);
                v16bf b0 = frag_cast(fb0), b1 = frag_cast(fb1);
                acc[0][0] = __builtin_amdgcn_wmma_f32_16x16x32_bf16(false, a0, false, b0,
                                    (short)0, acc[0][0], false, false);
                acc[0][1] = __builtin_amdgcn_wmma_f32_16x16x32_bf16(false, a0, false, b1,
                                    (short)0, acc[0][1], false, false);
                acc[1][0] = __builtin_amdgcn_wmma_f32_16x16x32_bf16(false, a1, false, b0,
                                    (short)0, acc[1][0], false, false);
                acc[1][1] = __builtin_amdgcn_wmma_f32_16x16x32_bf16(false, a1, false, b1,
                                    (short)0, acc[1][1], false, false);
            }
        }
    }

    // Epilogue: D layout -> VGPR r: lanes 0-15 M=r, lanes 16-31 M=r+8; N = lane&15.
    #pragma unroll
    for (int mt = 0; mt < 2; ++mt) {
        #pragma unroll
        for (int nt = 0; nt < 2; ++nt) {
            int hrow = H0 + wave_n * 4 + nt * 2 + pr;
            int ob = o0 + wave_m * 32 + mt * 16 + hi * 8;
            size_t base = (((size_t)b * COUT + ob) * HH + hrow) * WW + (W0 + pc);
            #pragma unroll
            for (int r = 0; r < 8; ++r)
                out[base + (size_t)r * (HH * WW)] = acc[mt][nt][r];
        }
    }
}

// ---------------------------------------------------------------------------
extern "C" void kernel_launch(void* const* d_in, const int* in_sizes, int n_in,
                              void* d_out, int out_size, void* d_ws, size_t ws_size,
                              hipStream_t stream) {
    (void)in_sizes; (void)n_in; (void)out_size; (void)ws_size;
    const float* x    = (const float*)d_in[0];
    const float* fc_w = (const float*)d_in[1];
    const float* ch_w = (const float*)d_in[2];
    const float* ch_b = (const float*)d_in[3];
    const float* fl_w = (const float*)d_in[4];
    const float* fl_b = (const float*)d_in[5];
    const float* sp_w = (const float*)d_in[6];
    const float* sp_b = (const float*)d_in[7];
    const float* ex_w = (const float*)d_in[8];
    const float* ex_b = (const float*)d_in[9];
    const float* weight = (const float*)d_in[10];
    float* out = (float*)d_out;

    char* base = (char*)d_ws;
    size_t off = 0;
    auto take = [&](size_t bytes) -> char* {
        char* p = base + off;
        off = (off + bytes + 255) & ~(size_t)255;
        return p;
    };
    float* pooled = (float*)take((size_t)BATCH * CIN * 4);
    float* ch     = (float*)take((size_t)BATCH * CIN * 4);
    float* fl     = (float*)take((size_t)BATCH * COUT * 4);
    float* sp     = (float*)take((size_t)BATCH * 9 * 4);
    float* ex     = (float*)take((size_t)BATCH * 4 * 4);
    size_t weff_bytes = (size_t)BATCH * 9 * COUT * CIN * 2;       // 37.75 MB
    size_t xcl_bytes  = (size_t)BATCH * HP * HP * CIN * 2;        // 55.12 MB
    unsigned short* weff = (unsigned short*)take(weff_bytes);
    unsigned short* xcl  = (unsigned short*)take(xcl_bytes);

    pool_kernel<<<BATCH * CIN, 256, 0, stream>>>(x, pooled);
    attn_kernel<<<BATCH, 256, 0, stream>>>(pooled, fc_w, ch_w, ch_b, fl_w, fl_b,
                                           sp_w, sp_b, ex_w, ex_b, ch, fl, sp, ex);
    weff_kernel<<<dim3(COUT, BATCH), 256, 0, stream>>>(weight, ch, fl, sp, ex, weff);
    (void)hipMemsetAsync(xcl, 0, xcl_bytes, stream);
    xpad_kernel<<<dim3(HH, CIN / 64, BATCH), 256, 0, stream>>>(x, xcl);
    conv_kernel<<<dim3(49, COUT / 128, BATCH), 256, 0, stream>>>(weff, xcl, out);
}
